// HierarchicalAttention_8143257993361
// MI455X (gfx1250) — matrix operand order repaired and checked by reading, compiled, not run on hardware
//
#include <hip/hip_runtime.h>

// ---------------- common types / helpers ----------------
typedef __attribute__((ext_vector_type(16))) __bf16 v16bf;
typedef __attribute__((ext_vector_type(8)))  float  v8f;

struct alignas(16) U4 { unsigned x, y, z, w; };
union FragBF { v16bf v; U4 q[2]; };

enum { NB = 2, SS = 2048, EE = 1024, HH = 16, DD = 64, MM = NB * SS };

static __device__ inline unsigned short f2bf(float f) {
  unsigned u = __float_as_uint(f);
  u += 0x7FFFu + ((u >> 16) & 1u);          // round-to-nearest-even
  return (unsigned short)(u >> 16);
}
static __device__ inline float bf2f(unsigned short h) {
  return __uint_as_float(((unsigned)h) << 16);
}
static __device__ inline unsigned pk2(float a, float b) {
  return (unsigned)f2bf(a) | ((unsigned)f2bf(b) << 16);
}
static __device__ inline v8f wmma_bf16(const v16bf& a, const v16bf& b, const v8f& c) {
  // D = A(16x32 bf16) * B(32x16 bf16) + C(16x16 f32)
  return __builtin_amdgcn_wmma_f32_16x16x32_bf16(false, a, false, b, (short)0, c, false, false);
}
static __device__ inline v8f zero8() {
  v8f z = {0.f, 0.f, 0.f, 0.f, 0.f, 0.f, 0.f, 0.f};
  return z;
}

// CDNA5 async global->LDS copy (ASYNCcnt tracked). INST_OFFSET is added to
// BOTH the LDS dest and the memory address (ISA 08_async_tensor §4.4), so one
// address pair serves consecutive 16-byte chunks.
template <int OFF>
static __device__ inline void async_ld16(unsigned lds_off, unsigned long long gaddr) {
  asm volatile("global_load_async_to_lds_b128 %0, %1, off offset:%c2"
               :: "v"(lds_off), "v"(gaddr), "n"(OFF) : "memory");
}
static __device__ inline void wait_async0() {
  asm volatile("s_wait_asynccnt 0" ::: "memory");
}
static __device__ inline unsigned lds_addr(const void* p) {
  return (unsigned)(unsigned long long)p;   // flat LDS addr: low 32 bits = LDS offset
}

// ---------------- generic WMMA GEMM:  out = A @ W + b ----------------
// WG tile: 128 (M) x 64 (N), K staged 32 at a time in LDS (bf16).
// 8 waves; wave w handles rows [w*16, w*16+16), all 4 N-subtiles of 16.
// A source selected at runtime: f32 (optionally row-gathered) or bf16
// (optionally concatenated from two buffers along K, each 1024 wide).
// bf16-A staging uses CDNA5 async global->LDS loads.
// Epilogue: store bf16, or gate-mix (sigmoid) into f32 d_out.
__global__ __launch_bounds__(256)
void hattn_gemm_kernel(const float* __restrict__ Af,          // f32 A (or null)
                       const unsigned short* __restrict__ Ab0,// bf16 A part0
                       const unsigned short* __restrict__ Ab1,// bf16 A part1 (K>=1024) or null
                       const int* __restrict__ gather,        // row indices into Af, or null
                       const float* __restrict__ W,           // [K][Nout] f32
                       const float* __restrict__ bias,        // [Nout] f32
                       unsigned short* __restrict__ outb,     // bf16 out (or null)
                       float* __restrict__ outf,              // f32 out for gate (or null)
                       const unsigned short* __restrict__ mix0,
                       const unsigned short* __restrict__ mix1,
                       int M, int K, int Nout, int lda)
{
  __shared__ alignas(16) unsigned short Alds[128][32];   // 8 KB, [m][k]
  __shared__ alignas(16) unsigned short Wlds[64][32];    // 4 KB, transposed [n][k]

  const int tid  = threadIdx.x;
  const int w    = tid >> 5;
  const int lane = tid & 31;
  const int lh   = lane & 15;
  const int sel  = lane >> 4;
  const int m0   = blockIdx.y * 128;
  const int nb   = blockIdx.x * 64;

  v8f acc[4];
#pragma unroll
  for (int c = 0; c < 4; ++c) acc[c] = zero8();

  for (int kk = 0; kk < K; kk += 32) {
    // ---- stage A tile (128 x 32 bf16) ----
    {
      const int row = tid >> 1;
      const int kh  = (tid & 1) * 16;
      const int mrow = m0 + row;
      if (Af) {
        const long arow = gather ? (long)gather[mrow] : (long)mrow;
        const float4* fp = (const float4*)(Af + arow * (long)lda + kk + kh);
        float4 f0 = fp[0], f1 = fp[1], f2 = fp[2], f3 = fp[3];
        U4 o0 = { pk2(f0.x, f0.y), pk2(f0.z, f0.w), pk2(f1.x, f1.y), pk2(f1.z, f1.w) };
        U4 o1 = { pk2(f2.x, f2.y), pk2(f2.z, f2.w), pk2(f3.x, f3.y), pk2(f3.z, f3.w) };
        *(U4*)&Alds[row][kh]     = o0;
        *(U4*)&Alds[row][kh + 8] = o1;
      } else {
        // pure bf16 copy -> CDNA5 async global->LDS path
        const unsigned short* base;
        if (Ab1 && kk >= 1024) base = Ab1 + (long)mrow * 1024 + (kk - 1024);
        else                   base = Ab0 + (long)mrow * 1024 + kk;
        const unsigned dst = lds_addr(&Alds[row][kh]);
        const unsigned long long ga = (unsigned long long)(base + kh);
        async_ld16<0>(dst, ga);
        async_ld16<16>(dst, ga);
      }
    }
    // ---- stage W tile transposed (32 x 64 -> [n][k]) ----
    {
      const int kr = tid >> 3;
      const int n8 = (tid & 7) * 8;
      const float* wrow = W + (long)(kk + kr) * Nout + nb + n8;
      const float4* wp = (const float4*)wrow;
      float4 w0 = wp[0], w1 = wp[1];
      Wlds[n8 + 0][kr] = f2bf(w0.x); Wlds[n8 + 1][kr] = f2bf(w0.y);
      Wlds[n8 + 2][kr] = f2bf(w0.z); Wlds[n8 + 3][kr] = f2bf(w0.w);
      Wlds[n8 + 4][kr] = f2bf(w1.x); Wlds[n8 + 5][kr] = f2bf(w1.y);
      Wlds[n8 + 6][kr] = f2bf(w1.z); Wlds[n8 + 7][kr] = f2bf(w1.w);
      if (kk + 32 < K)
        __builtin_prefetch(wrow + 32 * (long)Nout, 0, 1);   // next W tile
    }
    wait_async0();
    __syncthreads();

    // ---- A fragment (16x32): lane = row, K split per half-wave ----
    FragBF a;
    {
      const U4* ap = (const U4*)&Alds[w * 16 + lh][0];
      a.q[0] = ap[sel];       // K = sel*8 .. sel*8+7
      a.q[1] = ap[2 + sel];   // K = 16+sel*8 .. 16+sel*8+7
    }
    // ---- 4 N-subtiles: B fragment element e <-> K = sel*8+e (e<8), 16+sel*8+(e-8) (e>=8), col = lane&15 ----
#pragma unroll
    for (int c = 0; c < 4; ++c) {
      FragBF b;
      const U4* bp = (const U4*)&Wlds[c * 16 + lh][0];
      b.q[0] = bp[sel];
      b.q[1] = bp[2 + sel];
      acc[c] = wmma_bf16(a.v, b.v, acc[c]);
    }
    __syncthreads();
  }

  // ---- epilogue ----
#pragma unroll
  for (int c = 0; c < 4; ++c) {
    const int   ncol = nb + c * 16 + lh;
    const float bv   = bias[ncol];
#pragma unroll
    for (int r = 0; r < 8; ++r) {
      const int  mrow = m0 + w * 16 + r + sel * 8;
      const long oidx = (long)mrow * Nout + ncol;
      const float v   = acc[c][r] + bv;
      if (outf) {
        const float g  = 1.f / (1.f + __expf(-v));
        const float c0 = bf2f(mix0[oidx]);
        const float c1 = bf2f(mix1[oidx]);
        outf[oidx] = g * c0 + (1.f - g) * c1;
      } else {
        outb[oidx] = f2bf(v);
      }
    }
  }
}

// ---------------- causal flash attention (bf16 WMMA, f32 online softmax) ----
// Grid: (S/128, H, N). WG = 256 threads = 8 waves; wave w owns q rows
// [q0+w*16, +16). K/V streamed in 32-row blocks staged in LDS by the WG.
// K tile staged with CDNA5 async global->LDS; V transposed via VALU path.
__global__ __launch_bounds__(256)
void hattn_attn_kernel(const unsigned short* __restrict__ Q,
                       const unsigned short* __restrict__ Km,
                       const unsigned short* __restrict__ Vm,
                       unsigned short* __restrict__ O,
                       int Sdim, float scale)
{
  __shared__ alignas(16) unsigned short Klds[32][64];     // [k][d]   4 KB
  __shared__ alignas(16) unsigned short Vtlds[64][32];    // [d][k]   4 KB
  __shared__ alignas(16) unsigned short Plds[8][16][32];  // per-wave 8 KB

  const int tid  = threadIdx.x;
  const int w    = tid >> 5;
  const int lane = tid & 31;
  const int lh   = lane & 15;
  const int sel  = lane >> 4;
  const int q0   = blockIdx.x * 128;
  const int h    = blockIdx.y;
  const int n    = blockIdx.z;
  const int qw   = q0 + w * 16;

  // preload Q fragments for d-chunks [0,32) and [32,64)
  FragBF qf[2];
  {
    const unsigned short* qrow = Q + ((long)(n * Sdim + qw + lh)) * EE + h * DD;
    const U4* qp = (const U4*)qrow;
#pragma unroll
    for (int j = 0; j < 2; ++j) {
      qf[j].q[0] = qp[j * 4 + sel];
      qf[j].q[1] = qp[j * 4 + 2 + sel];
    }
  }

  float mrow[8], lrow[8];
  v8f oacc[4];
#pragma unroll
  for (int r = 0; r < 8; ++r) { mrow[r] = -3.0e38f; lrow[r] = 0.f; }
#pragma unroll
  for (int t = 0; t < 4; ++t) oacc[t] = zero8();

  const int nkb = (q0 + 128) / 32;
  for (int ib = 0; ib < nkb; ++ib) {
    const int kb = ib * 32;
    __syncthreads();
    // ---- cooperative stage of K (async -> LDS) and V (transposed) ----
    {
      const int r  = tid >> 3;
      const int c8 = (tid & 7) * 8;
      const long gbase = ((long)(n * Sdim + kb + r)) * EE + h * DD + c8;
      async_ld16<0>(lds_addr(&Klds[r][c8]), (unsigned long long)(Km + gbase));
      U4 vv = *(const U4*)(Vm + gbase);
      unsigned vr[4] = { vv.x, vv.y, vv.z, vv.w };
#pragma unroll
      for (int i = 0; i < 4; ++i) {
        Vtlds[c8 + 2 * i][r]     = (unsigned short)(vr[i] & 0xFFFFu);
        Vtlds[c8 + 2 * i + 1][r] = (unsigned short)(vr[i] >> 16);
      }
      if (ib + 1 < nkb) {                       // prefetch next K/V block
        __builtin_prefetch(Km + gbase + 32l * EE, 0, 1);
        __builtin_prefetch(Vm + gbase + 32l * EE, 0, 1);
      }
    }
    wait_async0();
    __syncthreads();

    if (kb <= qw + 15) {       // not fully masked for this wave
      // ---- S = Q K^T over d=64 (two chained WMMAs per 16-col subtile) ----
      v8f s0 = zero8(), s1 = zero8();
#pragma unroll
      for (int j = 0; j < 2; ++j) {
        FragBF b0, b1;
        const U4* kp0 = (const U4*)&Klds[lh][0];
        const U4* kp1 = (const U4*)&Klds[16 + lh][0];
        b0.q[0] = kp0[j * 4 + sel]; b0.q[1] = kp0[j * 4 + 2 + sel];
        b1.q[0] = kp1[j * 4 + sel]; b1.q[1] = kp1[j * 4 + 2 + sel];
        s0 = wmma_bf16(qf[j].v, b0.v, s0);
        s1 = wmma_bf16(qf[j].v, b1.v, s1);
      }

      // ---- online softmax (rows r+8*sel, cols lane&15 per C-layout) ----
#pragma unroll
      for (int r = 0; r < 8; ++r) {
        const int qr  = qw + r + sel * 8;
        const float t0 = (kb + lh      <= qr) ? s0[r] * scale : -3.0e38f;
        const float t1 = (kb + 16 + lh <= qr) ? s1[r] * scale : -3.0e38f;
        float mx = fmaxf(t0, t1);
        mx = fmaxf(mx, __shfl_xor(mx, 1));
        mx = fmaxf(mx, __shfl_xor(mx, 2));
        mx = fmaxf(mx, __shfl_xor(mx, 4));
        mx = fmaxf(mx, __shfl_xor(mx, 8));
        const float mnew  = fmaxf(mrow[r], mx);
        const float alpha = __expf(mrow[r] - mnew);
        const float p0 = __expf(t0 - mnew);
        const float p1 = __expf(t1 - mnew);
        float ps = p0 + p1;
        ps += __shfl_xor(ps, 1);
        ps += __shfl_xor(ps, 2);
        ps += __shfl_xor(ps, 4);
        ps += __shfl_xor(ps, 8);
        lrow[r] = lrow[r] * alpha + ps;
        mrow[r] = mnew;
#pragma unroll
        for (int t = 0; t < 4; ++t) oacc[t][r] *= alpha;
        Plds[w][r + sel * 8][lh]      = f2bf(p0);
        Plds[w][r + sel * 8][16 + lh] = f2bf(p1);
      }

      // ---- P (16x32) as A-fragment via per-wave LDS bounce ----
      FragBF pf;
      {
        const U4* pp = (const U4*)&Plds[w][lh][0];
        pf.q[0] = pp[sel];
        pf.q[1] = pp[2 + sel];
      }
      // ---- O += P @ V (4 d-subtiles) ----
#pragma unroll
      for (int t = 0; t < 4; ++t) {
        FragBF vf;
        const U4* vp = (const U4*)&Vtlds[t * 16 + lh][0];
        vf.q[0] = vp[sel];
        vf.q[1] = vp[2 + sel];
        oacc[t] = wmma_bf16(pf.v, vf.v, oacc[t]);
      }
    }
  }

  // ---- normalize and store bf16 ----
#pragma unroll
  for (int r = 0; r < 8; ++r) {
    const float inv = 1.0f / lrow[r];
    const long rowb = ((long)(n * Sdim + qw + r + sel * 8)) * EE + h * DD;
#pragma unroll
    for (int t = 0; t < 4; ++t)
      O[rowb + t * 16 + lh] = f2bf(oacc[t][r] * inv);
  }
}

// ---------------- launch ----------------
extern "C" void kernel_launch(void* const* d_in, const int* in_sizes, int n_in,
                              void* d_out, int out_size, void* d_ws, size_t ws_size,
                              hipStream_t stream) {
  (void)in_sizes; (void)n_in; (void)out_size; (void)ws_size;
  const float* values = (const float*)d_in[0];
  const float* keys   = (const float*)d_in[1];
  const float* query  = (const float*)d_in[2];
  const int*   cids   = (const int*)  d_in[3];
  const float* emb    = (const float*)d_in[4];
  const float* Wcv = (const float*)d_in[5],  *bcv = (const float*)d_in[6];
  const float* Wck = (const float*)d_in[7],  *bck = (const float*)d_in[8];
  const float* Wcq = (const float*)d_in[9],  *bcq = (const float*)d_in[10];
  const float* Wco = (const float*)d_in[11], *bco = (const float*)d_in[12];
  const float* Wpv = (const float*)d_in[13], *bpv = (const float*)d_in[14];
  const float* Wpk = (const float*)d_in[15], *bpk = (const float*)d_in[16];
  const float* Wpq = (const float*)d_in[17], *bpq = (const float*)d_in[18];
  const float* Wpo = (const float*)d_in[19], *bpo = (const float*)d_in[20];
  const float* Wg  = (const float*)d_in[21], *bg  = (const float*)d_in[22];
  float* out = (float*)d_out;

  const long SZ = (long)MM * EE;                        // elements per buffer
  unsigned short* bufV  = (unsigned short*)d_ws;
  unsigned short* bufK  = bufV  + SZ;
  unsigned short* bufQ  = bufK  + SZ;
  unsigned short* bufA  = bufQ  + SZ;
  unsigned short* charO = bufA  + SZ;
  unsigned short* concO = charO + SZ;

  const dim3 blk(256);
  const dim3 gg(EE / 64, MM / 128);        // (16, 32)
  const dim3 ga(SS / 128, HH, NB);         // (16, 16, 2)
  const float scale = 0.03125f;            // 1/sqrt(1024)

  // ---- char branch ----
  hattn_gemm_kernel<<<gg, blk, 0, stream>>>(values, nullptr, nullptr, nullptr,
      Wcv, bcv, bufV, nullptr, nullptr, nullptr, MM, EE, EE, EE);
  hattn_gemm_kernel<<<gg, blk, 0, stream>>>(keys, nullptr, nullptr, nullptr,
      Wck, bck, bufK, nullptr, nullptr, nullptr, MM, EE, EE, EE);
  hattn_gemm_kernel<<<gg, blk, 0, stream>>>(query, nullptr, nullptr, nullptr,
      Wcq, bcq, bufQ, nullptr, nullptr, nullptr, MM, EE, EE, EE);
  hattn_attn_kernel<<<ga, blk, 0, stream>>>(bufQ, bufK, bufV, bufA, SS, scale);
  hattn_gemm_kernel<<<gg, blk, 0, stream>>>(nullptr, bufA, nullptr, nullptr,
      Wco, bco, charO, nullptr, nullptr, nullptr, MM, EE, EE, EE);

  // ---- concept branch (A = emb[concept_ids] fused gather) ----
  hattn_gemm_kernel<<<gg, blk, 0, stream>>>(emb, nullptr, nullptr, cids,
      Wpv, bpv, bufV, nullptr, nullptr, nullptr, MM, EE, EE, EE);
  hattn_gemm_kernel<<<gg, blk, 0, stream>>>(emb, nullptr, nullptr, cids,
      Wpk, bpk, bufK, nullptr, nullptr, nullptr, MM, EE, EE, EE);
  hattn_gemm_kernel<<<gg, blk, 0, stream>>>(query, nullptr, nullptr, nullptr,
      Wpq, bpq, bufQ, nullptr, nullptr, nullptr, MM, EE, EE, EE);
  hattn_attn_kernel<<<ga, blk, 0, stream>>>(bufQ, bufK, bufV, bufA, SS, scale);
  hattn_gemm_kernel<<<gg, blk, 0, stream>>>(nullptr, bufA, nullptr, nullptr,
      Wpo, bpo, concO, nullptr, nullptr, nullptr, MM, EE, EE, EE);

  // ---- gate: sigmoid(concat(char,concept) @ Wg + bg) mix -> f32 out ----
  hattn_gemm_kernel<<<gg, blk, 0, stream>>>(nullptr, charO, concO, nullptr,
      Wg, bg, nullptr, out, charO, concO, MM, 2 * EE, EE, EE);
}
